// MultiHead_74311524155835
// MI455X (gfx1250) — matrix operand order repaired and checked by reading, not compiled, not run on hardware
//
#include <hip/hip_runtime.h>

typedef __attribute__((ext_vector_type(16))) _Float16 v16h;
typedef __attribute__((ext_vector_type(8)))  _Float16 v8h;
typedef __attribute__((ext_vector_type(8)))  float    v8f;

#define HEADS 6
#define QD    64
#define EMB   384
#define TT    512
#define BB    64

// ---- helpers ------------------------------------------------------------

static __device__ __forceinline__ v8h cvt8(const float* __restrict__ p) {
    float4 a = *(const float4*)p;
    float4 b = *(const float4*)(p + 4);
    v8h r;
    r[0] = (_Float16)a.x; r[1] = (_Float16)a.y; r[2] = (_Float16)a.z; r[3] = (_Float16)a.w;
    r[4] = (_Float16)b.x; r[5] = (_Float16)b.y; r[6] = (_Float16)b.z; r[7] = (_Float16)b.w;
    return r;
}

static __device__ __forceinline__ v16h join(v8h a, v8h b) {
    v16h r;
#pragma unroll
    for (int i = 0; i < 8; ++i) { r[i] = a[i]; r[8 + i] = b[i]; }
    return r;
}

static __device__ __forceinline__ v8h ld8(const _Float16* __restrict__ p) {
    return *(const v8h*)p;
}

static __device__ __forceinline__ v8f wmma(v16h a, v16h b, v8f c) {
    return __builtin_amdgcn_wmma_f32_16x16x32_f16(false, a, false, b, (short)0, c, false, false);
}

// ---- kernel 0: one-shot f32 -> f16 conversion (memory-bound, runs once) --
__global__ __launch_bounds__(256) void k_cvt(const float* __restrict__ src,
                                             _Float16* __restrict__ dst, int n8) {
    const int i = blockIdx.x * blockDim.x + threadIdx.x;  // one thread = 8 elems
    if (i < n8) *(v8h*)(dst + (size_t)i * 8) = cvt8(src + (size_t)i * 8);
}

// ---- kernel 1: fused QKV projection (y = x @ W.T), pure-f16 inner loop ---
// grid: (BT/128, EMB/64, 3), block 256 (8 waves). Wave tile: 16M x 64N.
// z==2 (V) stores transposed [B,H,D,T] so PxV B-fragments are contiguous.
__global__ __launch_bounds__(256) void k_qkv(const _Float16* __restrict__ xh,
                                             const _Float16* __restrict__ Wqh,
                                             const _Float16* __restrict__ Wkh,
                                             const _Float16* __restrict__ Wvh,
                                             _Float16* __restrict__ qb,
                                             _Float16* __restrict__ kbuf,
                                             _Float16* __restrict__ vtb) {
    const int w = threadIdx.x >> 5, lane = threadIdx.x & 31;
    const int lo = lane & 15, hi = lane >> 4;
    const int mb = (blockIdx.x * 8 + w) * 16;
    const int nb = blockIdx.y * 64;
    const int z  = blockIdx.z;
    const _Float16* __restrict__ W = (z == 0) ? Wqh : ((z == 1) ? Wkh : Wvh);

    v8f acc[4] = {};
    const _Float16* __restrict__ xr = xh + (size_t)(mb + lo) * EMB;  // A row = lane lo

#pragma unroll 4
    for (int kb = 0; kb < EMB; kb += 32) {
        v16h a = join(ld8(xr + kb + hi * 8), ld8(xr + kb + 16 + hi * 8));
#pragma unroll
        for (int nt = 0; nt < 4; ++nt) {
            const _Float16* wr = W + (size_t)(nb + nt * 16 + lo) * EMB + kb + hi * 8;
            acc[nt] = wmma(a, join(ld8(wr), ld8(wr + 16)), acc[nt]);
        }
    }

    if (z < 2) {
        _Float16* __restrict__ dst = z ? kbuf : qb;
#pragma unroll
        for (int nt = 0; nt < 4; ++nt) {
            const int n = nb + nt * 16 + lo;
            const int hh = n >> 6, d = n & 63;
#pragma unroll
            for (int i = 0; i < 8; ++i) {
                const int mm = mb + i + 8 * hi;
                const int b = mm >> 9, t = mm & 511;
                dst[(((size_t)b * HEADS + hh) * TT + t) * QD + d] = (_Float16)acc[nt][i];
            }
        }
    } else {
        const int b = mb >> 9;
        const int t0 = (mb & 511) + 8 * hi;  // rows are consecutive t -> contiguous store
#pragma unroll
        for (int nt = 0; nt < 4; ++nt) {
            const int n = nb + nt * 16 + lo;
            const int hh = n >> 6, d = n & 63;
            v8h pk;
#pragma unroll
            for (int i = 0; i < 8; ++i) pk[i] = (_Float16)acc[nt][i];
            *(v8h*)(vtb + (((size_t)b * HEADS + hh) * QD + d) * TT + t0) = pk;
        }
    }
}

// ---- kernel 2: flash attention, one wave per 16-row q tile ---------------
// grid: (T/16/8, B*H), block 256 (8 waves), 8KB LDS for P relayout.
__global__ __launch_bounds__(256) void k_attn(const _Float16* __restrict__ qbuf,
                                              const _Float16* __restrict__ kbuf,
                                              const _Float16* __restrict__ vtbuf,
                                              _Float16* __restrict__ obuf) {
    __shared__ _Float16 plds[8][16 * 32];

    const int w = threadIdx.x >> 5, lane = threadIdx.x & 31;
    const int lo = lane & 15, hi = lane >> 4;
    const int qt = blockIdx.x * 8 + w;
    const int bh = blockIdx.y;

    const _Float16* __restrict__ qbase  = qbuf  + (size_t)bh * TT * QD;
    const _Float16* __restrict__ kbase  = kbuf  + (size_t)bh * TT * QD;
    const _Float16* __restrict__ vtbase = vtbuf + (size_t)bh * QD * TT;

    // Q A-fragments (scale 1/sqrt(64)=0.125 folded in; exact in f16)
    const _Float16* qp = qbase + (size_t)(qt * 16 + lo) * QD;
    v8h q00 = ld8(qp + hi * 8),      q01 = ld8(qp + 16 + hi * 8);
    v8h q10 = ld8(qp + 32 + hi * 8), q11 = ld8(qp + 48 + hi * 8);
#pragma unroll
    for (int i = 0; i < 8; ++i) {
        const _Float16 s = (_Float16)0.125f;
        q00[i] *= s; q01[i] *= s; q10[i] *= s; q11[i] *= s;
    }
    const v16h qa0 = join(q00, q01);  // d = 0..31
    const v16h qa1 = join(q10, q11);  // d = 32..63

    v8f o0 = {}, o1 = {}, o2 = {}, o3 = {};
    float mrow[8], lrow[8];
#pragma unroll
    for (int i = 0; i < 8; ++i) { mrow[i] = -3.0e38f; lrow[i] = 0.0f; }

    const int qrb = qt * 16;
    _Float16* __restrict__ pr = plds[w];

    for (int kb = 0; kb < (qt + 1) * 16; kb += 32) {
        // S = Q @ K^T : two 16x16 tiles (columns kb..kb+15, kb+16..kb+31)
        const _Float16* kp0 = kbase + (size_t)(kb + lo) * QD;
        const _Float16* kp1 = kbase + (size_t)(kb + 16 + lo) * QD;
        v8f s0 = {}, s1 = {};
        s0 = wmma(qa0, join(ld8(kp0 + hi * 8), ld8(kp0 + hi * 8 + 16)), s0);
        s0 = wmma(qa1, join(ld8(kp0 + 32 + hi * 8), ld8(kp0 + 48 + hi * 8)), s0);
        s1 = wmma(qa0, join(ld8(kp1 + hi * 8), ld8(kp1 + hi * 8 + 16)), s1);
        s1 = wmma(qa1, join(ld8(kp1 + 32 + hi * 8), ld8(kp1 + 48 + hi * 8)), s1);

        if (kb + 31 > qrb) {  // diagonal-straddling block: causal mask
#pragma unroll
            for (int i = 0; i < 8; ++i) {
                const int qg = qrb + i + 8 * hi;
                if (kb + lo > qg)      s0[i] = -3.0e38f;
                if (kb + 16 + lo > qg) s1[i] = -3.0e38f;
            }
        }

        // online softmax (row = (vgpr i, half-wave); cols live across 16 lanes)
#pragma unroll
        for (int i = 0; i < 8; ++i) {
            float mx = fmaxf(s0[i], s1[i]);
#pragma unroll
            for (int off = 1; off < 16; off <<= 1) mx = fmaxf(mx, __shfl_xor(mx, off, 16));
            const float mnew  = fmaxf(mrow[i], mx);
            const float alpha = __expf(mrow[i] - mnew);
            mrow[i] = mnew;
            const float p0 = __expf(s0[i] - mnew);
            const float p1 = __expf(s1[i] - mnew);
            float rs = p0 + p1;
#pragma unroll
            for (int off = 1; off < 16; off <<= 1) rs += __shfl_xor(rs, off, 16);
            lrow[i] = lrow[i] * alpha + rs;
            o0[i] *= alpha; o1[i] *= alpha; o2[i] *= alpha; o3[i] *= alpha;
            const int r = i + 8 * hi;
            pr[r * 32 + lo]      = (_Float16)p0;
            pr[r * 32 + 16 + lo] = (_Float16)p1;
        }

        // re-layout P (C/D layout -> A layout) through LDS; same-wave LDS is
        // in-order, explicit CDNA5 split-counter wait stops compiler reorder too
        asm volatile("s_wait_dscnt 0" ::: "memory");
        const v16h pa = join(*(const v8h*)(pr + lo * 32 + hi * 8),
                             *(const v8h*)(pr + lo * 32 + 16 + hi * 8));

        // O += P @ V : V^T layout makes B-fragments two contiguous 16B loads per lane
        const _Float16* vp = vtbase + (size_t)lo * TT + kb + hi * 8;
        o0 = wmma(pa, join(ld8(vp),           ld8(vp + 16)),           o0);
        o1 = wmma(pa, join(ld8(vp + 16 * TT), ld8(vp + 16 * TT + 16)), o1);
        o2 = wmma(pa, join(ld8(vp + 32 * TT), ld8(vp + 32 * TT + 16)), o2);
        o3 = wmma(pa, join(ld8(vp + 48 * TT), ld8(vp + 48 * TT + 16)), o3);
    }

    // normalize and store O as f16 [B,H,T,D]
    _Float16* __restrict__ op = obuf + ((size_t)bh * TT + qt * 16) * QD;
#pragma unroll
    for (int i = 0; i < 8; ++i) {
        const float inv = 1.0f / lrow[i];
        const int r = i + 8 * hi;
        op[(size_t)r * QD + lo]      = (_Float16)(o0[i] * inv);
        op[(size_t)r * QD + 16 + lo] = (_Float16)(o1[i] * inv);
        op[(size_t)r * QD + 32 + lo] = (_Float16)(o2[i] * inv);
        op[(size_t)r * QD + 48 + lo] = (_Float16)(o3[i] * inv);
    }
}

// ---- kernel 3: output projection out = O @ Wp.T + bp (f32 out) -----------
__global__ __launch_bounds__(256) void k_proj(const _Float16* __restrict__ ob,
                                              const _Float16* __restrict__ Wph,
                                              const float* __restrict__ bp,
                                              float* __restrict__ out) {
    const int w = threadIdx.x >> 5, lane = threadIdx.x & 31;
    const int lo = lane & 15, hi = lane >> 4;
    const int mb = (blockIdx.x * 8 + w) * 16;
    const int nb = blockIdx.y * 64;

    const int m = mb + lo;            // A row
    const int b = m >> 9, t = m & 511;

    v8f acc[4] = {};
#pragma unroll 4
    for (int kb = 0; kb < EMB; kb += 32) {
        const int k0 = kb + hi * 8, k1 = kb + 16 + hi * 8;  // each 8-group within one head
        const _Float16* p0 = ob + (((size_t)b * HEADS + (k0 >> 6)) * TT + t) * QD + (k0 & 63);
        const _Float16* p1 = ob + (((size_t)b * HEADS + (k1 >> 6)) * TT + t) * QD + (k1 & 63);
        v16h a = join(ld8(p0), ld8(p1));
#pragma unroll
        for (int nt = 0; nt < 4; ++nt) {
            const _Float16* wr = Wph + (size_t)(nb + nt * 16 + lo) * EMB + kb + hi * 8;
            acc[nt] = wmma(a, join(ld8(wr), ld8(wr + 16)), acc[nt]);
        }
    }

#pragma unroll
    for (int nt = 0; nt < 4; ++nt) {
        const int n = nb + nt * 16 + lo;
        const float bias = bp[n];
#pragma unroll
        for (int i = 0; i < 8; ++i) {
            const int mm = mb + i + 8 * hi;
            out[(size_t)mm * EMB + n] = acc[nt][i] + bias;
        }
    }
}

// ---- launcher ------------------------------------------------------------
extern "C" void kernel_launch(void* const* d_in, const int* in_sizes, int n_in,
                              void* d_out, int out_size, void* d_ws, size_t ws_size,
                              hipStream_t stream) {
    const float* x  = (const float*)d_in[0];
    const float* Wq = (const float*)d_in[1];
    const float* Wk = (const float*)d_in[2];
    const float* Wv = (const float*)d_in[3];
    const float* Wp = (const float*)d_in[4];
    const float* bp = (const float*)d_in[5];
    float* out = (float*)d_out;

    const size_t N  = (size_t)BB * HEADS * TT * QD;  // 12.58M (= B*T*C too)
    const size_t NW = (size_t)EMB * EMB;             // 147456
    _Float16* qb  = (_Float16*)d_ws;
    _Float16* kb  = qb  + N;
    _Float16* vtb = kb  + N;   // V stored transposed [B,H,D,T]
    _Float16* obf = vtb + N;
    _Float16* xh  = obf + N;
    _Float16* wqh = xh  + N;
    _Float16* wkh = wqh + NW;
    _Float16* wvh = wkh + NW;
    _Float16* wph = wvh + NW;

    dim3 blk(256);
    // one-shot f32->f16 conversions (memory-bound, ~26 MB total)
    k_cvt<<<dim3((unsigned)((N / 8 + 255) / 256)),  blk, 0, stream>>>(x,  xh,  (int)(N / 8));
    k_cvt<<<dim3((unsigned)((NW / 8 + 255) / 256)), blk, 0, stream>>>(Wq, wqh, (int)(NW / 8));
    k_cvt<<<dim3((unsigned)((NW / 8 + 255) / 256)), blk, 0, stream>>>(Wk, wkh, (int)(NW / 8));
    k_cvt<<<dim3((unsigned)((NW / 8 + 255) / 256)), blk, 0, stream>>>(Wv, wvh, (int)(NW / 8));
    k_cvt<<<dim3((unsigned)((NW / 8 + 255) / 256)), blk, 0, stream>>>(Wp, wph, (int)(NW / 8));

    k_qkv <<<dim3(BB * TT / 128, EMB / 64, 3), blk, 0, stream>>>(xh, wqh, wkh, wvh, qb, kb, vtb);
    k_attn<<<dim3(TT / 16 / 8, BB * HEADS),    blk, 0, stream>>>(qb, kb, vtb, obf);
    k_proj<<<dim3(BB * TT / 128, EMB / 64),    blk, 0, stream>>>(obf, wph, bp, out);
}
